// MultiHeadLatentAttention_60378650247249
// MI455X (gfx1250) — matrix operand order repaired and checked by reading, not compiled, not run on hardware
//
#include <hip/hip_runtime.h>
#include <hip/hip_bf16.h>

typedef unsigned short u16;
typedef __attribute__((ext_vector_type(16))) __bf16 v16bf;
typedef __attribute__((ext_vector_type(8)))  float  v8f;

union FragBF { v16bf v; uint4 q[2]; };

#if __has_builtin(__builtin_amdgcn_global_load_async_to_lds_b128) && \
    __has_builtin(__builtin_amdgcn_s_wait_asynccnt)
#define HAVE_ASYNC_LDS 1
typedef int v4i __attribute__((vector_size(16)));
typedef __attribute__((address_space(1))) v4i as1_v4i;
typedef __attribute__((address_space(3))) v4i as3_v4i;
#endif

__device__ __forceinline__ void cp16_async(const u16* g, u16* l) {
#ifdef HAVE_ASYNC_LDS
  __builtin_amdgcn_global_load_async_to_lds_b128(
      (as1_v4i*)(uintptr_t)g,
      (as3_v4i*)(unsigned)(uintptr_t)l,
      0, 0);
#else
  *(uint4*)l = *(const uint4*)g;
#endif
}
__device__ __forceinline__ void cp_wait() {
#ifdef HAVE_ASYNC_LDS
  __builtin_amdgcn_s_wait_asynccnt(0);
#endif
}

__device__ __forceinline__ u16 f2bf(float f) {
  unsigned u = __float_as_uint(f);
  u += 0x7fffu + ((u >> 16) & 1u);            // round-to-nearest-even
  return (u16)(u >> 16);
}
__device__ __forceinline__ float bf2f(u16 h) {
  return __uint_as_float(((unsigned)h) << 16);
}
__device__ __forceinline__ v8f wmma_bf16(v16bf a, v16bf b, v8f c) {
  return __builtin_amdgcn_wmma_f32_16x16x32_bf16(false, a, false, b, (short)0, c, false, false);
}

#define SEQ 2048
#define DM  2048
#define NH  16
#define DH  128
#define DKV 512

// ---------------------------------------------------------------- elementwise
__global__ void convert_f32_bf16_kernel(const float* __restrict__ x,
                                        u16* __restrict__ y, int n) {
  int i = blockIdx.x * blockDim.x + threadIdx.x;
  if (i < n) y[i] = f2bf(x[i]);
}

// W[K][N] f32 -> Wt[N][K] bf16
__global__ void transpose_w_kernel(const float* __restrict__ W,
                                   u16* __restrict__ Wt, int K, int N) {
  long i = (long)blockIdx.x * blockDim.x + threadIdx.x;
  if (i >= (long)K * N) return;
  int k = (int)(i / N), n = (int)(i % N);
  Wt[(size_t)n * K + k] = f2bf(W[i]);
}

// ---------------------------------------------------------------- rmsnorm (f32 in -> bf16 out)
__global__ void rmsnorm_kernel(const float* __restrict__ X, const float* __restrict__ g,
                               u16* __restrict__ Y, int D) {
  int row  = blockIdx.x * (blockDim.x >> 5) + (threadIdx.x >> 5);
  int lane = threadIdx.x & 31;
  const float* x = X + (size_t)row * D;
  float ss = 0.f;
  for (int i = lane; i < D; i += 32) { float v = x[i]; ss += v * v; }
  #pragma unroll
  for (int off = 16; off; off >>= 1) ss += __shfl_xor(ss, off, 32);
  float s = rsqrtf(ss / (float)D + 1e-5f);
  u16* y = Y + (size_t)row * D;
  for (int i = lane; i < D; i += 32) y[i] = f2bf(x[i] * s * g[i]);
}

// ---------------------------------------------------------------- WMMA GEMM
// A[M][K] bf16 row-major, Bt[N][K] bf16 (B transposed), out f32 (Cf) or bf16 (Cb).
// Block: 128 threads = 4 waves; wave tile 16(M) x 64(N); grid (M/64, N/64).
// B tile (64 n-rows x 32 k) is shared by all 4 waves: staged in LDS via async
// copy (GLOBAL_LOAD_ASYNC_TO_LDS_B128, ASYNCcnt), double-buffered.
__global__ void gemm_wmma_kernel(const u16* __restrict__ A, const u16* __restrict__ Bt,
                                 float* __restrict__ Cf, u16* __restrict__ Cb,
                                 const float* __restrict__ bias,
                                 int M, int N, int K) {
  __shared__ __align__(16) u16 Blds[2][64][32];
  int tid  = threadIdx.x;
  int wave = tid >> 5, lane = tid & 31;
  int half = lane >> 4, lm = lane & 15;
  int m0 = blockIdx.x * 64 + wave * 16;
  int n0 = blockIdx.y * 64;

  const u16* arow  = A + (size_t)(m0 + lm) * K;
  const u16* bbase = Bt + (size_t)n0 * K;

  // cooperative B-tile copy: 256 chunks of 16B; thread handles chunks tid, tid+128
  int nl0 = tid >> 2,         kc0 = (tid & 3) * 8;
  int nl1 = (tid + 128) >> 2, kc1 = ((tid + 128) & 3) * 8;

  // prologue: issue tile 0
  cp16_async(bbase + (size_t)nl0 * K + kc0, &Blds[0][nl0][kc0]);
  cp16_async(bbase + (size_t)nl1 * K + kc1, &Blds[0][nl1][kc1]);

  v8f acc[4] = {};
  int nk = K >> 5;
  for (int ki = 0; ki < nk; ++ki) {
    int k = ki << 5;
    cp_wait();          // tile ki resident (this wave's chunks)
    __syncthreads();    // all waves' chunks resident; prev buffer free to overwrite
    if (ki + 1 < nk) {
      int kn = k + 32, nb = (ki + 1) & 1;
      cp16_async(bbase + (size_t)nl0 * K + kn + kc0, &Blds[nb][nl0][kc0]);
      cp16_async(bbase + (size_t)nl1 * K + kn + kc1, &Blds[nb][nl1][kc1]);
    }
    FragBF a;
    a.q[0] = *(const uint4*)(arow + k + half * 16);
    a.q[1] = *(const uint4*)(arow + k + half * 16 + 8);
    __builtin_prefetch(arow + k + 128, 0, 0);
    const u16 (*Bb)[32] = Blds[ki & 1];
    FragBF b[4];
    #pragma unroll
    for (int nt = 0; nt < 4; ++nt) {
      const u16* bp = &Bb[nt * 16 + lm][half * 16];
      b[nt].q[0] = *(const uint4*)bp;
      b[nt].q[1] = *(const uint4*)(bp + 8);
    }
    #pragma unroll
    for (int nt = 0; nt < 4; ++nt)
      acc[nt] = wmma_bf16(a.v, b[nt].v, acc[nt]);
  }
  #pragma unroll
  for (int nt = 0; nt < 4; ++nt) {
    int n = n0 + nt * 16 + lm;
    float bv = bias ? bias[n] : 0.f;
    #pragma unroll
    for (int r = 0; r < 8; ++r) {
      int m = m0 + r + 8 * half;
      float v = acc[nt][r] + bv;
      if (Cf) Cf[(size_t)m * N + n] = v;
      else    Cb[(size_t)m * N + n] = f2bf(v);
    }
  }
}

// ---------------------------------------------------------------- assemble Q/K/Vt with RoPE
// qb,qr,kb,kr: [4096][1024] bf16 ; v: [4096][2048] bf16
// Q,Km: [bh][s][128] bf16 ; Vt: [bh][128][s] bf16
__global__ void assemble_qkv_kernel(const u16* __restrict__ qb, const u16* __restrict__ qr,
                                    const u16* __restrict__ kb, const u16* __restrict__ kr,
                                    const u16* __restrict__ v,
                                    u16* __restrict__ Q, u16* __restrict__ Km,
                                    u16* __restrict__ Vt) {
  int t = blockIdx.x;
  int b = t >> 11, s = t & (SEQ - 1);
  for (int d = threadIdx.x; d < DM; d += blockDim.x) {
    int h = d >> 7, j = d & 127;
    int bh = b * NH + h;
    size_t qk = ((size_t)bh * SEQ + s) * DH + j;
    float qv, kv;
    if (j < 64) {
      qv = bf2f(qb[(size_t)t * 1024 + h * 64 + j]);
      kv = bf2f(kb[(size_t)t * 1024 + h * 64 + j]);
    } else {
      int j2 = j - 64, i = j2 & 31;
      float freq = __expf(-(float)i * 0.14391157f);   // ln(10000)/64
      float ang  = (float)s * freq;
      float fac  = __cosf(ang) + __sinf(ang);         // faithful: x*cos + x*sin
      qv = bf2f(qr[(size_t)t * 1024 + h * 64 + j2]) * fac;
      kv = bf2f(kr[(size_t)t * 1024 + h * 64 + j2]) * fac;
    }
    Q[qk]  = f2bf(qv);
    Km[qk] = f2bf(kv);
    Vt[((size_t)bh * DH + j) * SEQ + s] = v[(size_t)t * DM + d];
  }
}

// ---------------------------------------------------------------- flash attention (WMMA)
// One wave = one 16-row query tile of one (b,h). Block = 4 waves, grid (32, 32).
// O: token-major [4096][2048] bf16.
__global__ void attn_wmma_kernel(const u16* __restrict__ Q, const u16* __restrict__ Km,
                                 const u16* __restrict__ Vt, u16* __restrict__ O) {
  __shared__ __align__(16) u16 Plds[4][16][32];
  int wave = threadIdx.x >> 5, lane = threadIdx.x & 31;
  int half = lane >> 4, lm = lane & 15;
  int bh = blockIdx.y;
  int qt = blockIdx.x * 4 + wave;               // 0..127
  const u16* Qh = Q  + (size_t)bh * SEQ * DH;
  const u16* Kh = Km + (size_t)bh * SEQ * DH;
  const u16* Vh = Vt + (size_t)bh * DH * SEQ;

  // Q fragments: 4 chunks over d (A-matrix 16x32 layout)
  FragBF qf[4];
  const u16* qrow = Qh + (size_t)(qt * 16 + lm) * DH;
  #pragma unroll
  for (int c = 0; c < 4; ++c) {
    qf[c].q[0] = *(const uint4*)(qrow + c * 32 + half * 16);
    qf[c].q[1] = *(const uint4*)(qrow + c * 32 + half * 16 + 8);
  }

  v8f acc[8] = {};
  float mrow[8], lrow[8];
  #pragma unroll
  for (int r = 0; r < 8; ++r) { mrow[r] = -3.0e38f; lrow[r] = 0.f; }
  const float scale = 0.08838834764831845f;     // 1/sqrt(128)

  int nb = (qt * 16 + 16 + 31) >> 5;            // 32-key blocks (causal)
  for (int kb = 0; kb < nb; ++kb) {
    // S = Q K^T for 32 keys -> two 16x16 C tiles; batch d-chunks in pairs for ILP
    v8f s0 = {}, s1 = {};
    #pragma unroll
    for (int c = 0; c < 4; c += 2) {
      FragBF b00, b01, b10, b11;
      const u16* k00 = Kh + (size_t)(kb * 32 + lm) * DH + c * 32 + half * 16;
      const u16* k01 = Kh + (size_t)(kb * 32 + 16 + lm) * DH + c * 32 + half * 16;
      b00.q[0] = *(const uint4*)k00;        b00.q[1] = *(const uint4*)(k00 + 8);
      b01.q[0] = *(const uint4*)k01;        b01.q[1] = *(const uint4*)(k01 + 8);
      b10.q[0] = *(const uint4*)(k00 + 32); b10.q[1] = *(const uint4*)(k00 + 40);
      b11.q[0] = *(const uint4*)(k01 + 32); b11.q[1] = *(const uint4*)(k01 + 40);
      s0 = wmma_bf16(qf[c].v,     b00.v, s0);
      s1 = wmma_bf16(qf[c].v,     b01.v, s1);
      s0 = wmma_bf16(qf[c + 1].v, b10.v, s0);
      s1 = wmma_bf16(qf[c + 1].v, b11.v, s1);
    }
    // online softmax; C layout: row = r + 8*half, col = lm (+16)
    float alpha[8], p0[8], p1[8];
    #pragma unroll
    for (int r = 0; r < 8; ++r) {
      int qrg  = qt * 16 + r + 8 * half;
      int key0 = kb * 32 + lm, key1 = key0 + 16;
      float v0 = (key0 <= qrg) ? s0[r] * scale : -3.0e38f;
      float v1 = (key1 <= qrg) ? s1[r] * scale : -3.0e38f;
      float mx = fmaxf(v0, v1);
      #pragma unroll
      for (int off = 1; off < 16; off <<= 1) mx = fmaxf(mx, __shfl_xor(mx, off, 16));
      float mnew = fmaxf(mrow[r], mx);
      float a = __expf(mrow[r] - mnew);
      p0[r] = __expf(v0 - mnew);
      p1[r] = __expf(v1 - mnew);
      float rs = p0[r] + p1[r];
      #pragma unroll
      for (int off = 1; off < 16; off <<= 1) rs += __shfl_xor(rs, off, 16);
      lrow[r] = lrow[r] * a + rs;
      mrow[r] = mnew;
      alpha[r] = a;
    }
    #pragma unroll
    for (int j = 0; j < 8; ++j)
      #pragma unroll
      for (int r = 0; r < 8; ++r) acc[j][r] *= alpha[r];

    // P (C layout f32) -> LDS bf16 -> A-fragment layout
    #pragma unroll
    for (int r = 0; r < 8; ++r) {
      Plds[wave][r + 8 * half][lm]      = f2bf(p0[r]);
      Plds[wave][r + 8 * half][lm + 16] = f2bf(p1[r]);
    }
    asm volatile("s_wait_dscnt 0" ::: "memory");
    FragBF pf;
    pf.q[0] = *(const uint4*)&Plds[wave][lm][half * 16];
    pf.q[1] = *(const uint4*)&Plds[wave][lm][half * 16 + 8];

    // O += P @ V  (Vt gives contiguous B fragments)
    #pragma unroll
    for (int j = 0; j < 8; ++j) {
      FragBF bv;
      const u16* vp = Vh + (size_t)(j * 16 + lm) * SEQ + kb * 32 + half * 16;
      bv.q[0] = *(const uint4*)vp; bv.q[1] = *(const uint4*)(vp + 8);
      acc[j] = wmma_bf16(pf.v, bv.v, acc[j]);
    }
  }

  // epilogue: divide by l, store token-major bf16 [4096][2048]
  int b = bh >> 4, h = bh & (NH - 1);
  #pragma unroll
  for (int j = 0; j < 8; ++j) {
    #pragma unroll
    for (int r = 0; r < 8; ++r) {
      int srow = qt * 16 + r + 8 * half;
      int tok  = b * SEQ + srow;
      int col  = h * DH + j * 16 + lm;
      O[(size_t)tok * DM + col] = f2bf(acc[j][r] / lrow[r]);
    }
  }
}

// ---------------------------------------------------------------- host launcher
extern "C" void kernel_launch(void* const* d_in, const int* in_sizes, int n_in,
                              void* d_out, int out_size, void* d_ws, size_t ws_size,
                              hipStream_t stream) {
  const float* x     = (const float*)d_in[0];
  const float* W_dq  = (const float*)d_in[1];
  const float* W_uq  = (const float*)d_in[2];
  const float* W_dkv = (const float*)d_in[3];
  const float* W_uk  = (const float*)d_in[4];
  const float* W_uv  = (const float*)d_in[5];
  const float* W_qr  = (const float*)d_in[6];
  const float* W_kr  = (const float*)d_in[7];
  const float* g_q   = (const float*)d_in[8];
  const float* g_kv  = (const float*)d_in[9];
  const float* W_out = (const float*)d_in[10];
  const float* b_out = (const float*)d_in[11];
  float* outp = (float*)d_out;

  const int M = 2 * SEQ;  // 4096 tokens
  char* ws = (char*)d_ws;
  size_t off = 0;
  auto alloc = [&](size_t bytes) { size_t o = off; off = (off + bytes + 255) & ~(size_t)255; return o; };

  size_t o_xb    = alloc((size_t)M * DM * 2);
  size_t o_wdq   = alloc((size_t)DKV * DM * 2);
  size_t o_wuq   = alloc((size_t)1024 * DKV * 2);
  size_t o_wdkv  = alloc((size_t)DKV * DM * 2);
  size_t o_wuk   = alloc((size_t)1024 * DKV * 2);
  size_t o_wuv   = alloc((size_t)DM * DKV * 2);
  size_t o_wqr   = alloc((size_t)1024 * DKV * 2);
  size_t o_wkr   = alloc((size_t)1024 * DM * 2);
  size_t o_wout  = alloc((size_t)DM * DM * 2);
  size_t o_cqf   = alloc((size_t)M * DKV * 4);   // also reused as O (bf16) later
  size_t o_ckvf  = alloc((size_t)M * DKV * 4);
  size_t o_cqb   = alloc((size_t)M * DKV * 2);
  size_t o_ckvb  = alloc((size_t)M * DKV * 2);
  size_t o_qb    = alloc((size_t)M * 1024 * 2);
  size_t o_qr    = alloc((size_t)M * 1024 * 2);
  size_t o_kb    = alloc((size_t)M * 1024 * 2);
  size_t o_kr    = alloc((size_t)M * 1024 * 2);
  size_t o_v     = alloc((size_t)M * DM * 2);
  size_t o_Q     = alloc((size_t)2 * NH * SEQ * DH * 2);
  size_t o_K     = alloc((size_t)2 * NH * SEQ * DH * 2);
  size_t o_Vt    = alloc((size_t)2 * NH * DH * SEQ * 2);
  size_t o_O     = o_cqf;                        // reuse dead c_q/c_kv f32 region (16MB)
  (void)ws_size; (void)in_sizes; (void)n_in; (void)out_size;

  u16* xb = (u16*)(ws + o_xb);

  // 1) x -> bf16
  {
    int n = M * DM;
    convert_f32_bf16_kernel<<<(n + 255) / 256, 256, 0, stream>>>(x, xb, n);
  }
  // 2) transpose weights to bf16 [N][K]
  struct WT { const float* w; size_t o; int K, N; } wts[8] = {
    { W_dq,  o_wdq,  DM,  DKV  }, { W_uq,  o_wuq,  DKV, 1024 },
    { W_dkv, o_wdkv, DM,  DKV  }, { W_uk,  o_wuk,  DKV, 1024 },
    { W_uv,  o_wuv,  DKV, DM   }, { W_qr,  o_wqr,  DKV, 1024 },
    { W_kr,  o_wkr,  DM,  1024 }, { W_out, o_wout, DM,  DM   },
  };
  for (int i = 0; i < 8; ++i) {
    long n = (long)wts[i].K * wts[i].N;
    transpose_w_kernel<<<(unsigned)((n + 255) / 256), 256, 0, stream>>>(
        wts[i].w, (u16*)(ws + wts[i].o), wts[i].K, wts[i].N);
  }
  auto gemm = [&](size_t oA, size_t oB, float* Cf, u16* Cb, const float* bias,
                  int m, int n, int k) {
    dim3 g(m / 64, n / 64), blk(128);
    gemm_wmma_kernel<<<g, blk, 0, stream>>>((const u16*)(ws + oA), (const u16*)(ws + oB),
                                            Cf, Cb, bias, m, n, k);
  };
  // 3) down/rope projections from x
  gemm(o_xb, o_wdq,  (float*)(ws + o_cqf),  nullptr, nullptr, M, DKV,  DM);
  gemm(o_xb, o_wdkv, (float*)(ws + o_ckvf), nullptr, nullptr, M, DKV,  DM);
  gemm(o_xb, o_wkr,  nullptr, (u16*)(ws + o_kr), nullptr, M, 1024, DM);
  // 4) rmsnorm -> bf16
  rmsnorm_kernel<<<M / 8, 256, 0, stream>>>((const float*)(ws + o_cqf),  g_q,
                                            (u16*)(ws + o_cqb),  DKV);
  rmsnorm_kernel<<<M / 8, 256, 0, stream>>>((const float*)(ws + o_ckvf), g_kv,
                                            (u16*)(ws + o_ckvb), DKV);
  // 5) up projections
  gemm(o_cqb,  o_wuq, nullptr, (u16*)(ws + o_qb), nullptr, M, 1024, DKV);
  gemm(o_cqb,  o_wqr, nullptr, (u16*)(ws + o_qr), nullptr, M, 1024, DKV);
  gemm(o_ckvb, o_wuk, nullptr, (u16*)(ws + o_kb), nullptr, M, 1024, DKV);
  gemm(o_ckvb, o_wuv, nullptr, (u16*)(ws + o_v),  nullptr, M, DM,   DKV);
  // 6) assemble Q/K (rope) and Vt
  assemble_qkv_kernel<<<M, 256, 0, stream>>>(
      (const u16*)(ws + o_qb), (const u16*)(ws + o_qr),
      (const u16*)(ws + o_kb), (const u16*)(ws + o_kr),
      (const u16*)(ws + o_v),
      (u16*)(ws + o_Q), (u16*)(ws + o_K), (u16*)(ws + o_Vt));
  // 7) attention
  {
    dim3 g(SEQ / 16 / 4, 2 * NH), blk(128);
    attn_wmma_kernel<<<g, blk, 0, stream>>>((const u16*)(ws + o_Q), (const u16*)(ws + o_K),
                                            (const u16*)(ws + o_Vt), (u16*)(ws + o_O));
  }
  // 8) output projection + bias -> f32 d_out
  gemm(o_O, o_wout, outp, nullptr, b_out, M, DM, DM);
}
